// EnhancedGNNAutoencoder_8890582302923
// MI455X (gfx1250) — matrix operand order repaired and checked by reading, not compiled, not run on hardware
//
#include <hip/hip_runtime.h>

// ---------------------------------------------------------------------------
// GAT autoencoder for MI455X (gfx1250, wave32, WMMA).
// GEMMs use v_wmma_f32_16x16x32_bf16; segment softmax / scatter use L2 atomics.
// ---------------------------------------------------------------------------

typedef __attribute__((ext_vector_type(16))) __bf16 v16bf;
typedef __attribute__((ext_vector_type(8)))  __bf16 v8bf;
typedef __attribute__((ext_vector_type(8)))  float  v8f;

// float -> bf16 with round-to-nearest-even (bit-level, no __bf16 arithmetic)
__device__ __forceinline__ unsigned short f2bf(float f) {
    unsigned u = __float_as_uint(f);
    unsigned r = u + 0x7FFFu + ((u >> 16) & 1u);
    return (unsigned short)(r >> 16);
}

// order-preserving mapping float -> uint for atomicMax-based segment max
__device__ __forceinline__ unsigned enc_f(float f) {
    unsigned u = __float_as_uint(f);
    return (u & 0x80000000u) ? ~u : (u | 0x80000000u);
}
__device__ __forceinline__ float dec_f(unsigned u) {
    return (u & 0x80000000u) ? __uint_as_float(u & 0x7FFFFFFFu)
                             : __uint_as_float(~u);
}

// --------------------------- conversion / packing ---------------------------

__global__ void gat_cvt_bf16(const float* __restrict__ x,
                             unsigned short* __restrict__ xbf, int n) {
    int i = blockIdx.x * blockDim.x + threadIdx.x;
    if (i < n) xbf[i] = f2bf(x[i]);
}

// Wpack[(h*dout + n)*(din/2) + p] = {bf16(W[h][2p][n]), bf16(W[h][2p+1][n])}
__global__ void gat_pack_w(const float* __restrict__ W,
                           unsigned* __restrict__ wpack,
                           int H, int din, int dout) {
    int idx = blockIdx.x * blockDim.x + threadIdx.x;
    int halfk = din >> 1;
    int total = H * dout * halfk;
    if (idx >= total) return;
    int p    = idx % halfk;
    int rest = idx / halfk;
    int n    = rest % dout;
    int h    = rest / dout;
    float w0 = W[((size_t)h * din + 2 * p)     * dout + n];
    float w1 = W[((size_t)h * din + 2 * p + 1) * dout + n];
    wpack[idx] = (unsigned)f2bf(w0) | ((unsigned)f2bf(w1) << 16);
}

// ------------------------------- WMMA GEMM ----------------------------------
// xp[(n*H + h)*dout + o] = sum_k x[n][k] * W[h][k][o]
// One wave per 16x16 output tile; K loop in steps of 32 (bf16 WMMA).
__global__ void __launch_bounds__(32)
gat_gemm_xp(const unsigned short* __restrict__ xbf,
            const unsigned* __restrict__ wpack,
            float* __restrict__ xp,
            int N, int din, int dout, int H) {
    const int lane = threadIdx.x & 31;
    const int half = lane >> 4;     // K-half selector (this code's order: k = 16*half + s)
    const int l16  = lane & 15;
    const int mtile = blockIdx.x;
    const int h     = blockIdx.y;
    const int ntile = blockIdx.z;

    int row = mtile * 16 + l16;         // A row this lane supplies
    int rowc = (row < N) ? row : 0;     // clamp (stores are guarded)
    const int ncol = ntile * 16 + l16;  // B column this lane supplies

    const __bf16* Abase = (const __bf16*)xbf;
    const __bf16* Bbase = (const __bf16*)wpack;
    const int kblocks = din >> 5;
    const int halfk = din >> 1;

    v8f acc = {};
    for (int kb = 0; kb < kblocks; ++kb) {
        // A: row-major bf16, two contiguous 8-elem chunks (16B each)
        const __bf16* ap = Abase + (size_t)rowc * din + kb * 32 + half * 16;
        v8bf alo = *(const v8bf*)ap;
        v8bf ahi = *(const v8bf*)(ap + 8);
        v16bf A;
#pragma unroll
        for (int i = 0; i < 8; ++i) { A[i] = alo[i]; A[i + 8] = ahi[i]; }

        // B: 8 contiguous dwords (16 bf16) from the packed weights
        const __bf16* bp = Bbase +
            2 * ((size_t)(h * dout + ncol) * halfk + kb * 16 + half * 8);
        v16bf B = *(const v16bf*)bp;

        acc = __builtin_amdgcn_wmma_f32_16x16x32_bf16(
            false, A, false, B, (short)0, acc, false, false);
    }

    // D layout: lane L, VGPR j -> M = j + 8*half, Nc = l16
    const int mbase = mtile * 16 + 8 * half;
#pragma unroll
    for (int j = 0; j < 8; ++j) {
        int m = mbase + j;
        if (m < N)
            xp[((size_t)m * H + h) * dout + ntile * 16 + l16] = acc[j];
    }
}

// --------------------------- attention pipeline -----------------------------

__global__ void gat_scores(const float* __restrict__ xp,
                           const float* __restrict__ a_src,
                           const float* __restrict__ a_dst,
                           float* __restrict__ s_src,
                           float* __restrict__ s_dst,
                           int NH, int H, int dout) {
    int idx = blockIdx.x * blockDim.x + threadIdx.x;
    if (idx >= NH) return;
    int h = idx % H;
    const float* rowp = xp + (size_t)idx * dout;
    float s1 = 0.f, s2 = 0.f;
    for (int o = 0; o < dout; ++o) {
        float v = rowp[o];
        s1 += v * a_src[h * dout + o];
        s2 += v * a_dst[h * dout + o];
    }
    s_src[idx] = s1;
    s_dst[idx] = s2;
}

__global__ void gat_init(float* __restrict__ accum, float* __restrict__ denom,
                         unsigned* __restrict__ menc, int NH, int NHD) {
    int idx = blockIdx.x * blockDim.x + threadIdx.x;
    if (idx < NHD) accum[idx] = 0.f;
    if (idx < NH) { denom[idx] = 0.f; menc[idx] = 0u; }
}

__global__ void gat_edge_max(const int* __restrict__ src,
                             const int* __restrict__ dst,
                             int E, int Etot,
                             const float* __restrict__ s_src,
                             const float* __restrict__ s_dst,
                             unsigned* __restrict__ menc, int H) {
    int t = blockIdx.x * blockDim.x + threadIdx.x;
    if (t >= Etot) return;
    int s = (t < E) ? src[t] : (t - E);   // self-loops appended
    int d = (t < E) ? dst[t] : (t - E);
    for (int h = 0; h < H; ++h) {
        float x = s_src[s * H + h] + s_dst[d * H + h];
        float el = (x > 0.f) ? x : 0.2f * x;   // leaky_relu(0.2)
        atomicMax(menc + d * H + h, enc_f(el));
    }
}

__global__ void gat_edge_exp(const int* __restrict__ src,
                             const int* __restrict__ dst,
                             int E, int Etot,
                             const float* __restrict__ s_src,
                             const float* __restrict__ s_dst,
                             const unsigned* __restrict__ menc,
                             float* __restrict__ ebuf,
                             float* __restrict__ denom, int H) {
    int t = blockIdx.x * blockDim.x + threadIdx.x;
    if (t >= Etot) return;
    int s = (t < E) ? src[t] : (t - E);
    int d = (t < E) ? dst[t] : (t - E);
    for (int h = 0; h < H; ++h) {
        float x = s_src[s * H + h] + s_dst[d * H + h];
        float el = (x > 0.f) ? x : 0.2f * x;
        float m = dec_f(menc[d * H + h]);
        float e = __expf(el - m);
        ebuf[t * H + h] = e;
        atomicAdd(denom + d * H + h, e);
    }
}

__global__ void gat_edge_scatter(const int* __restrict__ src,
                                 const int* __restrict__ dst,
                                 int E, int Etot,
                                 const float* __restrict__ ebuf,
                                 const float* __restrict__ denom,
                                 const float* __restrict__ xp,
                                 float* __restrict__ accum,
                                 int H, int dout) {
    int tid = blockIdx.x * blockDim.x + threadIdx.x;
    if (tid >= Etot * H) return;
    int t = tid / H, h = tid - t * H;
    int s = (t < E) ? src[t] : (t - E);
    int d = (t < E) ? dst[t] : (t - E);
    float w = ebuf[tid] / denom[d * H + h];
    const float4* xs = (const float4*)(xp + ((size_t)s * H + h) * dout);
    float* ac = accum + ((size_t)d * H + h) * dout;
    int q4 = dout >> 2;
    for (int q = 0; q < q4; ++q) {
        float4 v = xs[q];
        atomicAdd(ac + 4 * q + 0, w * v.x);
        atomicAdd(ac + 4 * q + 1, w * v.y);
        atomicAdd(ac + 4 * q + 2, w * v.z);
        atomicAdd(ac + 4 * q + 3, w * v.w);
    }
}

// head-mean + bias (+relu); write bf16 next-layer input, or f32 final output
__global__ void gat_finalize(const float* __restrict__ accum,
                             const float* __restrict__ bias,
                             unsigned short* __restrict__ xbf_next,
                             float* __restrict__ out_f32,
                             int N, int H, int dout, int do_relu, int write_f32) {
    int tid = blockIdx.x * blockDim.x + threadIdx.x;
    if (tid >= N * dout) return;
    int n = tid / dout, o = tid - n * dout;
    float a = 0.f;
    for (int h = 0; h < H; ++h)
        a += accum[((size_t)n * H + h) * dout + o];
    float v = a / (float)H + bias[o];
    if (do_relu) v = (v > 0.f) ? v : 0.f;
    if (write_f32) out_f32[tid] = v;
    else           xbf_next[tid] = f2bf(v);
}

// ------------------------------- launcher -----------------------------------

extern "C" void kernel_launch(void* const* d_in, const int* in_sizes, int n_in,
                              void* d_out, int out_size, void* d_ws, size_t ws_size,
                              hipStream_t stream) {
    const float* x  = (const float*)d_in[0];
    const int*   ei = (const int*)d_in[1];
    const int N = in_sizes[0] / 128;   // 20000
    const int E = in_sizes[1] / 2;     // 320000
    const int Etot = E + N;            // + self-loops
    const int* srcp = ei;
    const int* dstp = ei + E;

    // workspace carve-out (256B-aligned)
    char* base = (char*)d_ws;
    size_t off = 0;
    auto carve = [&](size_t bytes) {
        size_t o = off;
        off += (bytes + 255) & ~(size_t)255;
        return o;
    };
    unsigned short* xbf = (unsigned short*)(base + carve((size_t)N * 128 * 2));
    float* xp    = (float*)(base + carve((size_t)N * 512 * 4));   // max H*dout=512
    float* accum = (float*)(base + carve((size_t)N * 512 * 4));
    float* ssrc  = (float*)(base + carve((size_t)N * 4 * 4));
    float* sdst  = (float*)(base + carve((size_t)N * 4 * 4));
    unsigned* menc = (unsigned*)(base + carve((size_t)N * 4 * 4));
    float* denom = (float*)(base + carve((size_t)N * 4 * 4));
    float* ebuf  = (float*)(base + carve((size_t)Etot * 4 * 4));
    unsigned* wpack = (unsigned*)(base + carve((size_t)4 * 128 * 64 * 4));
    (void)ws_size; (void)n_in;

    // layer 0 input -> bf16
    {
        int n = N * 128;
        gat_cvt_bf16<<<(n + 255) / 256, 256, 0, stream>>>(x, xbf, n);
    }

    struct LL { int din, dout, H, relu; };
    const LL L[6] = {
        {128, 128, 4, 1}, {128, 64, 4, 1}, {64, 32, 4, 0},
        {32, 64, 1, 1},   {64, 128, 1, 1}, {128, 128, 1, 0},
    };

    for (int i = 0; i < 6; ++i) {
        const float* W  = (const float*)d_in[2 + 4 * i];
        const float* as = (const float*)d_in[3 + 4 * i];
        const float* ad = (const float*)d_in[4 + 4 * i];
        const float* bb = (const float*)d_in[5 + 4 * i];
        const int din = L[i].din, dout = L[i].dout, H = L[i].H;
        const int NH = N * H, NHD = N * H * dout;

        // pack weights for WMMA B layout
        {
            int n = H * dout * (din >> 1);
            gat_pack_w<<<(n + 255) / 256, 256, 0, stream>>>(W, wpack, H, din, dout);
        }
        // projection GEMM: one wave per 16x16 tile
        {
            dim3 g((N + 15) / 16, H, dout / 16);
            gat_gemm_xp<<<g, 32, 0, stream>>>(xbf, wpack, xp, N, din, dout, H);
        }
        gat_scores<<<(NH + 255) / 256, 256, 0, stream>>>(xp, as, ad, ssrc, sdst,
                                                         NH, H, dout);
        gat_init<<<(NHD + 255) / 256, 256, 0, stream>>>(accum, denom, menc, NH, NHD);
        gat_edge_max<<<(Etot + 255) / 256, 256, 0, stream>>>(
            srcp, dstp, E, Etot, ssrc, sdst, menc, H);
        gat_edge_exp<<<(Etot + 255) / 256, 256, 0, stream>>>(
            srcp, dstp, E, Etot, ssrc, sdst, menc, ebuf, denom, H);
        {
            int n = Etot * H;
            gat_edge_scatter<<<(n + 255) / 256, 256, 0, stream>>>(
                srcp, dstp, E, Etot, ebuf, denom, xp, accum, H, dout);
        }
        {
            int n = N * dout;
            int last = (i == 5);
            gat_finalize<<<(n + 255) / 256, 256, 0, stream>>>(
                accum, bb, xbf, (float*)d_out, N, H, dout, L[i].relu, last);
        }
    }
}
